// OrganizedScaleFusion_21028159881495
// MI455X (gfx1250) — matrix operand order repaired and checked by reading, not compiled, not run on hardware
//
#include <hip/hip_runtime.h>
#include <cstdint>
#include <cstddef>

// ---------------------------------------------------------------------------
// Problem dims (compile-time constants from the reference)
// ---------------------------------------------------------------------------
#define BB      4
#define SS      512
#define DD      1024
#define HH      16
#define HDIM    64
#define KR      64          // routes per query
#define MTOT    (BB * SS)   // 2048 rows for both GEMMs
#define N_QKV   (3 * DD)    // 3072
#define SCALE_C 0.125f      // 1/sqrt(64)

// ---------------------------------------------------------------------------
// bf16 helpers (storage-only; math in f32)
// ---------------------------------------------------------------------------
typedef __attribute__((ext_vector_type(8)))  __bf16 bf16x8;
typedef __attribute__((ext_vector_type(16))) __bf16 bf16x16;
typedef __attribute__((ext_vector_type(8)))  float  f32x8;

union Frag16 { bf16x16 f; bf16x8 h[2]; };

__device__ __forceinline__ float bf2f(unsigned short u) {
    return __uint_as_float(((unsigned int)u) << 16);
}
__device__ __forceinline__ unsigned short f2bf(float f) {
    unsigned int u = __float_as_uint(f);
    u += 0x7FFFu + ((u >> 16) & 1u);          // round-to-nearest-even
    return (unsigned short)(u >> 16);
}

// ---------------------------------------------------------------------------
// 1) f32 -> bf16 straight conversion (x)
// ---------------------------------------------------------------------------
__global__ __launch_bounds__(256) void conv_f32_bf16_kernel(
    const float* __restrict__ src, unsigned short* __restrict__ dst, int n) {
    int i = blockIdx.x * 256 + threadIdx.x;
    if (i < n) dst[i] = f2bf(src[i]);
}

// ---------------------------------------------------------------------------
// 2) transpose+convert: W [K,N] f32 -> WT [N,K] bf16 (so GEMM B-frags are
//    contiguous 16B loads along K)
// ---------------------------------------------------------------------------
__global__ __launch_bounds__(256) void tconv_kernel(
    const float* __restrict__ w, unsigned short* __restrict__ wT, int K, int N) {
    int i = blockIdx.x * 256 + threadIdx.x;
    if (i >= K * N) return;
    int k = i / N, n = i - k * N;             // read coalesced along N
    wT[(size_t)n * K + k] = f2bf(w[i]);
}

// ---------------------------------------------------------------------------
// 3) WMMA bf16 GEMM:  C[M,N] = A[M,K] * BT[N,K]^T + bias
//    Wave tile 32x64 = 2(M) x 4(N) v_wmma_f32_16x16x32_bf16 tiles:
//    8 WMMAs per 12 x 16B fragment loads per K-step (~21 flops/byte).
//    8 waves arranged 4(M) x 2(N) -> 128x128 block tile.
//    EPI==0: scatter into q/k/v bf16 buffers laid out [B,H,S,HD]
//    EPI==1: plain f32 store to Cf[M,N]
// ---------------------------------------------------------------------------
template <int EPI>
__global__ __launch_bounds__(256) void wmma_gemm_kernel(
    const unsigned short* __restrict__ A,    // [M,K] bf16 row-major
    const unsigned short* __restrict__ BT,   // [N,K] bf16 row-major (B^T)
    const float* __restrict__ bias,          // [N]
    float* __restrict__ Cf,                  // EPI==1
    unsigned short* __restrict__ q_out,      // EPI==0
    unsigned short* __restrict__ k_out,
    unsigned short* __restrict__ v_out,
    int M, int N, int K) {
    const int tid  = threadIdx.x;
    const int wid  = tid >> 5;
    const int lane = tid & 31;
    const int lr   = lane & 15;               // row/col within 16-tile
    const int lg   = lane >> 4;               // lane group (K-half select)
    const int wm   = wid >> 1;                // 0..3
    const int wn   = wid & 1;                 // 0..1
    const int M0   = blockIdx.y * 128 + wm * 32;
    const int N0   = blockIdx.x * 128 + wn * 64;

    f32x8 acc[2][4] = {};

    for (int k0 = 0; k0 < K; k0 += 32) {
        Frag16 af[2], bf[4];
        // A fragment: lane holds row M0+mi*16+lr, K = k0+8*lg+{0..7} and +16
#pragma unroll
        for (int mi = 0; mi < 2; ++mi) {
            const unsigned short* p = A + (size_t)(M0 + mi * 16 + lr) * K + k0 + 8 * lg;
            af[mi].h[0] = *(const bf16x8*)(p);
            af[mi].h[1] = *(const bf16x8*)(p + 16);
        }
        // B fragment: lane holds column N0+ni*16+lr, K = k0+8*lg+{0..7} and +16
#pragma unroll
        for (int ni = 0; ni < 4; ++ni) {
            const unsigned short* p = BT + (size_t)(N0 + ni * 16 + lr) * K + k0 + 8 * lg;
            bf[ni].h[0] = *(const bf16x8*)(p);
            bf[ni].h[1] = *(const bf16x8*)(p + 16);
        }
        if (k0 + 32 < K) {   // gfx1250 global_prefetch_b8 for next K-step
            __builtin_prefetch(A  + (size_t)(M0 + lr) * K + k0 + 32, 0, 1);
            __builtin_prefetch(BT + (size_t)(N0 + lr) * K + k0 + 32, 0, 1);
        }
#pragma unroll
        for (int mi = 0; mi < 2; ++mi)
#pragma unroll
            for (int ni = 0; ni < 4; ++ni)
                acc[mi][ni] = __builtin_amdgcn_wmma_f32_16x16x32_bf16(
                    false, af[mi].f, false, bf[ni].f,
                    (short)0, acc[mi][ni], false, false);
    }

    // Epilogue. C/D layout: lane (lg,lr), VGPR i -> (m = 8*lg+i, n = lr)
#pragma unroll
    for (int mi = 0; mi < 2; ++mi)
#pragma unroll
        for (int ni = 0; ni < 4; ++ni)
#pragma unroll
            for (int i = 0; i < 8; ++i) {
                int m = M0 + mi * 16 + lg * 8 + i;
                int n = N0 + ni * 16 + lr;
                float val = acc[mi][ni][i] + bias[n];
                if (EPI == 1) {
                    Cf[(size_t)m * N + n] = val;
                } else {
                    // n = three*1024 + h*64 + hd ; m = b*512 + s
                    int three = n >> 10;
                    int h     = (n >> 6) & (HH - 1);
                    int hd    = n & (HDIM - 1);
                    int bb    = m >> 9;
                    int s     = m & (SS - 1);
                    size_t dst = (((size_t)bb * HH + h) * SS + s) * HDIM + hd;
                    unsigned short bv = f2bf(val);
                    if (three == 0)      q_out[dst] = bv;
                    else if (three == 1) k_out[dst] = bv;
                    else                 v_out[dst] = bv;
                }
            }
}

// ---------------------------------------------------------------------------
// 4) neighbor blend (route-independent):
//    kb[s] = 0.75*k[s] + 0.125*(k[max(s-1,0)] + k[min(s+1,S-1)]) ; same for v
// ---------------------------------------------------------------------------
__global__ __launch_bounds__(256) void blend_kernel(
    const unsigned short* __restrict__ k, const unsigned short* __restrict__ v,
    unsigned short* __restrict__ kb, unsigned short* __restrict__ vb) {
    int idx = blockIdx.x * 256 + threadIdx.x;           // over B*H*S*HD = 2^21
    if (idx >= BB * HH * SS * HDIM) return;
    int d  = idx & (HDIM - 1);
    int s  = (idx >> 6) & (SS - 1);
    int bh = idx >> 15;                                  // S*HD = 32768
    int sm = (s > 0) ? s - 1 : 0;
    int sp = (s < SS - 1) ? s + 1 : SS - 1;
    size_t base = (size_t)bh * (SS * HDIM);
    float kc = bf2f(k[base + (size_t)s  * HDIM + d]);
    float kl = bf2f(k[base + (size_t)sm * HDIM + d]);
    float kr = bf2f(k[base + (size_t)sp * HDIM + d]);
    kb[idx] = f2bf(0.75f * kc + 0.125f * (kl + kr));
    float vc = bf2f(v[base + (size_t)s  * HDIM + d]);
    float vl = bf2f(v[base + (size_t)sm * HDIM + d]);
    float vr = bf2f(v[base + (size_t)sp * HDIM + d]);
    vb[idx] = f2bf(0.75f * vc + 0.125f * (vl + vr));
}

// ---------------------------------------------------------------------------
// 5) routed attention: one wave (32 lanes) per query.
//    scores: 2 routes/lane, q broadcast via LDS; wave32 shfl softmax;
//    output: lane owns dims (lane, lane+32), coalesced gather of blended V.
//    Writes [B,S,H*HD] bf16 (GEMM2 A-layout).
// ---------------------------------------------------------------------------
__global__ __launch_bounds__(256) void routed_attn_kernel(
    const unsigned short* __restrict__ qb,   // [B,H,S,HD] bf16
    const unsigned short* __restrict__ kb,   // blended K
    const unsigned short* __restrict__ vb,   // blended V
    const int* __restrict__ routes,          // [S,KR]
    unsigned short* __restrict__ ao) {       // [B,S,H*HD] bf16
    __shared__ float lds_q[8][HDIM];
    __shared__ float lds_w[8][KR];
    __shared__ int   lds_r[8][KR];

    const int tid  = threadIdx.x;
    const int w    = tid >> 5;
    const int lane = tid & 31;
    const int gq   = blockIdx.x * 8 + w;     // 0 .. B*H*S-1
    const int b    = gq >> 13;               // / (H*S)
    const int rem  = gq & 8191;
    const int h    = rem >> 9;
    const int s    = rem & (SS - 1);

    const size_t bh = (size_t)b * HH + h;
    const unsigned short* qrow  = qb + (bh * SS + s) * HDIM;
    const unsigned short* kbase = kb + bh * SS * HDIM;
    const unsigned short* vbase = vb + bh * SS * HDIM;

    // stage q (f32) and routes in LDS
    lds_q[w][lane]      = bf2f(qrow[lane]);
    lds_q[w][lane + 32] = bf2f(qrow[lane + 32]);
    int r0 = routes[s * KR + lane];
    int r1 = routes[s * KR + lane + 32];
    lds_r[w][lane]      = r0;
    lds_r[w][lane + 32] = r1;
    __syncthreads();

    // scores for the 2 routes this lane owns
    const unsigned short* k0p = kbase + (size_t)r0 * HDIM;
    const unsigned short* k1p = kbase + (size_t)r1 * HDIM;
    float sc0 = 0.f, sc1 = 0.f;
#pragma unroll
    for (int d = 0; d < HDIM; ++d) {
        float qd = lds_q[w][d];
        sc0 += qd * bf2f(k0p[d]);
        sc1 += qd * bf2f(k1p[d]);
    }
    sc0 = (r0 > s) ? -1e9f : sc0 * SCALE_C;   // causal route mask
    sc1 = (r1 > s) ? -1e9f : sc1 * SCALE_C;

    // wave32 softmax over 64 scores (2/lane)
    float mx = fmaxf(sc0, sc1);
#pragma unroll
    for (int off = 16; off > 0; off >>= 1)
        mx = fmaxf(mx, __shfl_xor(mx, off, 32));
    float e0 = __expf(sc0 - mx);
    float e1 = __expf(sc1 - mx);
    float sum = e0 + e1;
#pragma unroll
    for (int off = 16; off > 0; off >>= 1)
        sum += __shfl_xor(sum, off, 32);
    float inv = 1.0f / sum;
    lds_w[w][lane]      = e0 * inv;
    lds_w[w][lane + 32] = e1 * inv;
    __syncthreads();

    // output: lane owns dims (lane, lane+32); V gather is lane-coalesced
    float acc0 = 0.f, acc1 = 0.f;
#pragma unroll
    for (int kk = 0; kk < KR; ++kk) {
        float wgt = lds_w[w][kk];
        int   rr  = lds_r[w][kk];
        const unsigned short* vp = vbase + (size_t)rr * HDIM;
        acc0 += wgt * bf2f(vp[lane]);
        acc1 += wgt * bf2f(vp[lane + 32]);
    }
    unsigned short* orow = ao + ((size_t)(b * SS + s) * DD) + h * HDIM;
    orow[lane]      = f2bf(acc0);
    orow[lane + 32] = f2bf(acc1);
}

// ---------------------------------------------------------------------------
// Host-side orchestration
// ---------------------------------------------------------------------------
extern "C" void kernel_launch(void* const* d_in, const int* in_sizes, int n_in,
                              void* d_out, int out_size, void* d_ws, size_t ws_size,
                              hipStream_t stream) {
    (void)in_sizes; (void)n_in; (void)out_size; (void)ws_size;
    const float* x      = (const float*)d_in[0];   // [B,S,D]
    const float* w_qkv  = (const float*)d_in[1];   // [D, 3D]
    const float* b_qkv  = (const float*)d_in[2];   // [3D]
    const float* w_out  = (const float*)d_in[3];   // [D, D]
    const float* b_out  = (const float*)d_in[4];   // [D]
    const int*   routes = (const int*)d_in[5];     // [S, KR]
    float* out = (float*)d_out;                    // [B,S,D] f32

    // workspace carve-out (bf16 buffers), 256B-aligned
    uint8_t* ws = (uint8_t*)d_ws;
    size_t off = 0;
    auto carve = [&](size_t bytes) -> void* {
        void* p = ws + off;
        off += (bytes + 255) & ~(size_t)255;
        return p;
    };
    unsigned short* xb     = (unsigned short*)carve((size_t)MTOT * DD * 2);      // x bf16 [M,K]
    unsigned short* wqkvT  = (unsigned short*)carve((size_t)N_QKV * DD * 2);     // [3D, D] bf16
    unsigned short* woutT  = (unsigned short*)carve((size_t)DD * DD * 2);        // [D, D] bf16
    unsigned short* qbuf   = (unsigned short*)carve((size_t)BB * HH * SS * HDIM * 2);
    unsigned short* kbuf   = (unsigned short*)carve((size_t)BB * HH * SS * HDIM * 2);
    unsigned short* vbuf   = (unsigned short*)carve((size_t)BB * HH * SS * HDIM * 2);
    unsigned short* kblend = (unsigned short*)carve((size_t)BB * HH * SS * HDIM * 2);
    unsigned short* vblend = (unsigned short*)carve((size_t)BB * HH * SS * HDIM * 2);
    unsigned short* aobuf  = (unsigned short*)carve((size_t)MTOT * DD * 2);      // attn out bf16 [M,K]

    // 1) convert x
    {
        int n = MTOT * DD;                                   // 2,097,152
        conv_f32_bf16_kernel<<<n / 256, 256, 0, stream>>>(x, xb, n);
    }
    // 2) transpose+convert weights
    {
        int n = DD * N_QKV;                                  // 3,145,728
        tconv_kernel<<<n / 256, 256, 0, stream>>>(w_qkv, wqkvT, DD, N_QKV);
    }
    {
        int n = DD * DD;                                     // 1,048,576
        tconv_kernel<<<n / 256, 256, 0, stream>>>(w_out, woutT, DD, DD);
    }
    // 3) QKV GEMM: [2048,1024] x [1024,3072] -> scatter q/k/v bf16
    {
        dim3 grid(N_QKV / 128, MTOT / 128);                  // (24, 16)
        wmma_gemm_kernel<0><<<grid, 256, 0, stream>>>(
            xb, wqkvT, b_qkv, nullptr, qbuf, kbuf, vbuf, MTOT, N_QKV, DD);
    }
    // 4) neighbor blend of K, V
    {
        int n = BB * HH * SS * HDIM;                         // 2,097,152
        blend_kernel<<<n / 256, 256, 0, stream>>>(kbuf, vbuf, kblend, vblend);
    }
    // 5) routed attention (one wave per query; 8 queries per block)
    {
        int nq = BB * HH * SS;                               // 32768
        routed_attn_kernel<<<nq / 8, 256, 0, stream>>>(qbuf, kblend, vblend, routes, aobuf);
    }
    // 6) output GEMM: [2048,1024] x [1024,1024] + b_out -> f32 d_out
    {
        dim3 grid(DD / 128, MTOT / 128);                     // (8, 16)
        wmma_gemm_kernel<1><<<grid, 256, 0, stream>>>(
            aobuf, woutT, b_out, out, nullptr, nullptr, nullptr, MTOT, DD, DD);
    }
}
